// SelfAttention_2173253452014
// MI455X (gfx1250) — matrix operand order repaired and checked by reading, not compiled, not run on hardware
//
#include <hip/hip_runtime.h>

// ---------------------------------------------------------------------------
// Self-attention (B=8, C=128, N=4096) for gfx1250, flash-attention style.
//   proj:    qT[b][n][c], kT[b][n][c], v[b][c][n] (f16) via v_wmma_f32_16x16x32_f16
//   attn:    split-K flash attention; S^T = K*Q^T per 64-key tile, online
//            softmax fully in registers, O += P*V with P converted in-register
//            to the WMMA A-layout.  K/V tiles staged to LDS with
//            global_load_async_to_lds_b128 (ASYNCcnt) + prefetch.
//   combine: merge split-K partials (m, l, unnormalized O) -> out.
// ---------------------------------------------------------------------------

typedef __attribute__((ext_vector_type(16))) _Float16 v16h;
typedef __attribute__((ext_vector_type(8)))  _Float16 v8h;
typedef __attribute__((ext_vector_type(8)))  float    v8f;

#define BATCH 8
#define CDIM  128
#define NDIM  4096
#define QBLK  128                 // queries per attention WG
#define NQB   (NDIM / QBLK)       // 32 query blocks per batch
#define LOG2E 1.44269504088896f

__device__ __forceinline__ v16h cat8(v8h lo, v8h hi) {
  return __builtin_shufflevector(lo, hi, 0,1,2,3,4,5,6,7,8,9,10,11,12,13,14,15);
}

// A-fragment (16x32 f16, M on lanes): lanes<16 hold K in {0..7}U{16..23},
// lanes>=16 hold {8..15}U{24..31}.  sub = (lane>>4)*8.
__device__ __forceinline__ v16h frag_a_f16(const _Float16* row, int koff, int sub) {
  v8h lo = *(const v8h*)(row + koff + sub);
  v8h hi = *(const v8h*)(row + koff + 16 + sub);
  return cat8(lo, hi);
}

// B-fragment (32x16 f16, N on lanes): lanes<16 hold K in {0..7}U{16..23},
// lanes>=16 hold {8..15}U{24..31}.  off = koff + (lane>>4)*8.
__device__ __forceinline__ v16h frag_b_f16(const _Float16* row, int off) {
  v8h lo = *(const v8h*)(row + off);
  v8h hi = *(const v8h*)(row + off + 16);
  return cat8(lo, hi);
}

__device__ __forceinline__ v8h cvt8_f32(const float* p) {
  v8f t;
  #pragma unroll
  for (int i = 0; i < 8; ++i) t[i] = p[i];
  return __builtin_convertvector(t, v8h);
}

__device__ __forceinline__ v16h frag_a_f32(const float* row, int koff, int sub) {
  return cat8(cvt8_f32(row + koff + sub), cvt8_f32(row + koff + 16 + sub));
}

__device__ __forceinline__ v8f wmma_f16(v16h a, v16h b, v8f c) {
  return __builtin_amdgcn_wmma_f32_16x16x32_f16(false, a, false, b, (short)0, c,
                                                false, false);
}

// Async DMA: 16B global -> LDS, tracked by ASYNCcnt (CDNA5).
// GVS mode: address = SADDR(u64) + VADDR(i32); vdst = LDS byte offset.
__device__ __forceinline__ void async_ld_b128(unsigned ldsoff, unsigned voff,
                                              unsigned long long saddr) {
  asm volatile("global_load_async_to_lds_b128 %0, %1, %2"
               :: "v"(ldsoff), "v"(voff), "s"(saddr) : "memory");
}
__device__ __forceinline__ void wait_asynccnt0() {
  asm volatile("s_wait_asynccnt 0x0" ::: "memory");
}
__device__ __forceinline__ unsigned lds_off(const void* p) {
  // low 32 bits of the flat shared-aperture address == LDS byte offset
  return (unsigned)(unsigned long long)p;
}

// ---------------------------------------------------------------------------
// Projection: q/k/v = W*x + b.  One WG per (batch, 64-pixel tile).
// Wave w owns output-channel tile d = 16w..16w+15.
// ---------------------------------------------------------------------------
__global__ __launch_bounds__(256) void proj_kernel(
    const float* __restrict__ x,
    const float* __restrict__ Wq, const float* __restrict__ bq,
    const float* __restrict__ Wk, const float* __restrict__ bk,
    const float* __restrict__ Wv, const float* __restrict__ bv,
    _Float16* __restrict__ qT, _Float16* __restrict__ kT,
    _Float16* __restrict__ vv) {
  constexpr int XPAD = 136;                 // halves per row, bank-conflict pad
  __shared__ _Float16 xT[64 * XPAD];        // x tile transposed: [n][c] f16

  const int b    = blockIdx.x / (NDIM / 64);
  const int n0   = (blockIdx.x % (NDIM / 64)) * 64;
  const int t    = threadIdx.x;
  const int lane = t & 31;
  const int w    = t >> 5;
  const int hf   = lane >> 4;
  const int lid  = lane & 15;

  {
    const int c    = t >> 1;
    const int noff = (t & 1) * 32;
    const float* xp = x + ((size_t)b * CDIM + c) * NDIM + n0 + noff;
    #pragma unroll
    for (int k = 0; k < 32; ++k)
      xT[(noff + k) * XPAD + c] = (_Float16)xp[k];
  }
  __syncthreads();

  const int sub = hf * 8;
  const int row_d = w * 16 + lid;

  v16h aq[4], ak[4], av[4];
  #pragma unroll
  for (int ch = 0; ch < 4; ++ch) {
    aq[ch] = frag_a_f32(Wq + (size_t)row_d * CDIM, ch * 32, sub);
    ak[ch] = frag_a_f32(Wk + (size_t)row_d * CDIM, ch * 32, sub);
    av[ch] = frag_a_f32(Wv + (size_t)row_d * CDIM, ch * 32, sub);
  }
  float bqr[8], bkr[8], bvr[8];
  #pragma unroll
  for (int r = 0; r < 8; ++r) {
    const int dr = w * 16 + hf * 8 + r;     // D-layout: M=d rows per lane half
    bqr[r] = bq[dr]; bkr[r] = bk[dr]; bvr[r] = bv[dr];
  }

  for (int nt = 0; nt < 4; ++nt) {
    const _Float16* xrow = &xT[(nt * 16 + lid) * XPAD];
    v16h bx[4];
    #pragma unroll
    for (int ch = 0; ch < 4; ++ch)
      bx[ch] = frag_b_f16(xrow, ch * 32 + hf * 8);

    v8f accq = {0,0,0,0,0,0,0,0}, acck = accq, accv = accq;
    #pragma unroll
    for (int ch = 0; ch < 4; ++ch) {
      accq = wmma_f16(aq[ch], bx[ch], accq);
      acck = wmma_f16(ak[ch], bx[ch], acck);
      accv = wmma_f16(av[ch], bx[ch], accv);
    }

    const int n_g = n0 + nt * 16 + lid;     // D-layout: N=n on lanes
    #pragma unroll
    for (int r = 0; r < 8; ++r) {
      const int d = w * 16 + hf * 8 + r;
      qT[((size_t)b * NDIM + n_g) * CDIM + d] = (_Float16)(accq[r] + bqr[r]);
      kT[((size_t)b * NDIM + n_g) * CDIM + d] = (_Float16)(acck[r] + bkr[r]);
      vv[((size_t)b * CDIM + d) * NDIM + n_g] = (_Float16)(accv[r] + bvr[r]);
    }
  }
}

// ---------------------------------------------------------------------------
// Split-K flash attention.  Grid = nsplit * BATCH * NQB blocks of 256 threads
// (8 waves).  Wave w owns 16 queries; each split covers tiles_per_split
// 64-key tiles and emits unnormalized O plus per-query (m, l).
// ---------------------------------------------------------------------------
__global__ __launch_bounds__(256) void attn_kernel(
    const _Float16* __restrict__ qT, const _Float16* __restrict__ kT,
    const _Float16* __restrict__ vv, float* __restrict__ Opart,
    float* __restrict__ Mpart, float* __restrict__ Lpart,
    int tiles_per_split) {
  constexpr int KPAD = 136, VPAD = 72;
  __shared__ _Float16 Ks[64 * KPAD];    // K tile:  [j][c]
  __shared__ _Float16 Vs[128 * VPAD];   // V tile:  [d][j]

  const int p    = blockIdx.x / (BATCH * NQB);
  const int rem  = blockIdx.x % (BATCH * NQB);
  const int b    = rem / NQB;
  const int qb   = rem % NQB;
  const int i0   = qb * QBLK;
  const int t    = threadIdx.x;
  const int lane = t & 31;
  const int w    = t >> 5;
  const int hf   = lane >> 4;
  const int lid  = lane & 15;

  // Q B-fragments, resident for the whole kernel (wave's 16 queries).
  const _Float16* qrow = qT + ((size_t)b * NDIM + i0 + w * 16 + lid) * CDIM;
  v16h qb4[4];
  #pragma unroll
  for (int ch = 0; ch < 4; ++ch)
    qb4[ch] = frag_b_f16(qrow, ch * 32 + hf * 8);

  const v8f vzero = {0,0,0,0,0,0,0,0};
  v8f O[8];
  #pragma unroll
  for (int dt = 0; dt < 8; ++dt) O[dt] = vzero;
  float m = -3.0e38f, l_sum = 0.0f;

  // Per-thread async-copy coordinates (K: 64 B/thread, V: 64 B/thread).
  const int jr = t >> 2, koff = (t & 3) * 32;          // K row / half-offset
  const int dr = t >> 1, voff = (t & 1) * 32;          // V row / half-offset
  const unsigned kdst = lds_off(&Ks[jr * KPAD + koff]);
  const unsigned vdst = lds_off(&Vs[dr * VPAD + voff]);
  const unsigned long long kbase = (unsigned long long)(kT + (size_t)b * NDIM * CDIM);
  const unsigned long long vbase = (unsigned long long)(vv + (size_t)b * CDIM * NDIM);

  const int kt0 = p * tiles_per_split;
  for (int kt = kt0; kt < kt0 + tiles_per_split; ++kt) {
    const int j0 = kt * 64;
    __syncthreads();                       // prior tile fully consumed
    {
      const unsigned kvo = (unsigned)(((j0 + jr) * CDIM + koff) * 2);
      const unsigned vvo = (unsigned)((dr * NDIM + j0 + voff) * 2);
      #pragma unroll
      for (int q4 = 0; q4 < 4; ++q4) {
        async_ld_b128(kdst + 16 * q4, kvo + 16 * q4, kbase);
        async_ld_b128(vdst + 16 * q4, vvo + 16 * q4, vbase);
      }
      if (kt + 1 < kt0 + tiles_per_split) {  // warm next tile into L2
        __builtin_prefetch(kT + ((size_t)b * NDIM + j0 + 64 + jr) * CDIM + koff, 0, 0);
        __builtin_prefetch(vv + ((size_t)b * CDIM + dr) * NDIM + j0 + 64 + voff, 0, 0);
      }
    }
    wait_asynccnt0();
    __syncthreads();                       // all waves' tiles landed

    // S^T = K * Q^T : four 16x16 tiles (j-local x i), D-layout: i on lanes.
    v8f St[4];
    #pragma unroll
    for (int jj = 0; jj < 4; ++jj) {
      const _Float16* krow = &Ks[(jj * 16 + lid) * KPAD];
      v8f acc = vzero;
      #pragma unroll
      for (int ch = 0; ch < 4; ++ch)
        acc = wmma_f16(frag_a_f16(krow, ch * 32, hf * 8), qb4[ch], acc);
      St[jj] = acc;
    }

    // Online softmax: each lane owns query i=lid; j lives in registers.
    float vmax = -3.0e38f;
    #pragma unroll
    for (int jj = 0; jj < 4; ++jj)
      #pragma unroll
      for (int r = 0; r < 8; ++r) vmax = fmaxf(vmax, St[jj][r]);
    vmax = fmaxf(vmax, __shfl_xor(vmax, 16, 32));     // merge lane halves
    const float mnew  = fmaxf(m, vmax);
    const float alpha = exp2f((m - mnew) * LOG2E);
    m = mnew;

    float rsum = 0.0f;
    v8h ph[4];
    #pragma unroll
    for (int jj = 0; jj < 4; ++jj)
      #pragma unroll
      for (int r = 0; r < 8; ++r) {
        const float pv = exp2f((St[jj][r] - m) * LOG2E);
        rsum += pv;
        ph[jj][r] = (_Float16)pv;
      }
    rsum  += __shfl_xor(rsum, 16, 32);
    l_sum  = l_sum * alpha + rsum;

    // Rescale O: alpha is per-query (on lanes); O rows need it per-VGPR.
    #pragma unroll
    for (int r = 0; r < 8; ++r) {
      const float ar = __shfl(alpha, hf * 8 + r, 32);
      #pragma unroll
      for (int dt = 0; dt < 8; ++dt) O[dt][r] *= ar;
    }

    // P already matches the WMMA A-fragment layout (ISA 16-bit A table).
    const v16h pa0 = cat8(ph[0], ph[1]);
    const v16h pa1 = cat8(ph[2], ph[3]);

    #pragma unroll
    for (int dt = 0; dt < 8; ++dt) {
      const _Float16* vrow = &Vs[(dt * 16 + lid) * VPAD];
      O[dt] = wmma_f16(pa0, frag_b_f16(vrow, hf * 8),      O[dt]);
      O[dt] = wmma_f16(pa1, frag_b_f16(vrow, 32 + hf * 8), O[dt]);
    }
  }

  // Emit partials: unnormalized O[d][i_local] plus per-query m, l.
  float* obase = Opart + ((((size_t)p * BATCH + b) * NQB + qb) * CDIM) * QBLK;
  #pragma unroll
  for (int r = 0; r < 8; ++r) {
    const int il = w * 16 + hf * 8 + r;
    #pragma unroll
    for (int dt = 0; dt < 8; ++dt) {
      const int d = dt * 16 + lid;              // D-layout: N=d on lanes
      obase[(size_t)d * QBLK + il] = O[dt][r];
    }
  }
  if (hf == 0) {
    const size_t sidx = (((size_t)p * BATCH + b) * NQB + qb) * QBLK + w * 16 + lid;
    Mpart[sidx] = m;
    Lpart[sidx] = l_sum;
  }
}

// ---------------------------------------------------------------------------
// Combine split-K partials: out[b][d][i] = sum_p O_p*e^(m_p-m*) / sum_p l_p*e^(m_p-m*)
// ---------------------------------------------------------------------------
__global__ __launch_bounds__(256) void combine_kernel(
    const float* __restrict__ Opart, const float* __restrict__ Mpart,
    const float* __restrict__ Lpart, float* __restrict__ out, int nsplit) {
  const size_t idx = (size_t)blockIdx.x * 256 + threadIdx.x;  // over B*C*N
  const int i  = (int)(idx % NDIM);
  const int d  = (int)((idx / NDIM) % CDIM);
  const int b  = (int)(idx / ((size_t)NDIM * CDIM));
  const int qb = i >> 7, il = i & (QBLK - 1);

  float mstar = -3.0e38f;
  for (int p = 0; p < nsplit; ++p) {
    const size_t s = (((size_t)p * BATCH + b) * NQB + qb) * QBLK + il;
    mstar = fmaxf(mstar, Mpart[s]);
  }
  float lsum = 0.0f, acc = 0.0f;
  for (int p = 0; p < nsplit; ++p) {
    const size_t s = (((size_t)p * BATCH + b) * NQB + qb) * QBLK + il;
    const float wgt = exp2f((Mpart[s] - mstar) * LOG2E);
    lsum += Lpart[s] * wgt;
    acc  += Opart[((((size_t)p * BATCH + b) * NQB + qb) * CDIM + d) * QBLK + il] * wgt;
  }
  out[idx] = acc / lsum;
}

// ---------------------------------------------------------------------------
extern "C" void kernel_launch(void* const* d_in, const int* in_sizes, int n_in,
                              void* d_out, int out_size, void* d_ws, size_t ws_size,
                              hipStream_t stream) {
  const float* x  = (const float*)d_in[0];
  const float* Wq = (const float*)d_in[1];
  const float* bq = (const float*)d_in[2];
  const float* Wk = (const float*)d_in[3];
  const float* bk = (const float*)d_in[4];
  const float* Wv = (const float*)d_in[5];
  const float* bv = (const float*)d_in[6];
  float* out = (float*)d_out;

  const size_t elems = (size_t)BATCH * NDIM * CDIM;      // 4M per tensor
  _Float16* qT = (_Float16*)d_ws;
  _Float16* kT = qT + elems;
  _Float16* vv = kT + elems;

  // Deterministic split count from workspace capacity.
  const size_t base      = 3 * elems * sizeof(_Float16);  // 24 MB for q/k/v
  const size_t o_split   = (size_t)BATCH * NQB * CDIM * QBLK * sizeof(float);
  const size_t ml_split  = 2 * (size_t)BATCH * NQB * QBLK * sizeof(float);
  int nsplit = 1;
  if (base + 4 * (o_split + ml_split) <= ws_size) nsplit = 4;
  else if (base + 2 * (o_split + ml_split) <= ws_size) nsplit = 2;

  float* Opart = (float*)((char*)d_ws + base);
  float* Mpart = Opart + (size_t)nsplit * BATCH * NQB * CDIM * QBLK;
  float* Lpart = Mpart + (size_t)nsplit * BATCH * NQB * QBLK;
  const int tiles_per_split = (NDIM / 64) / nsplit;

  proj_kernel<<<BATCH * (NDIM / 64), 256, 0, stream>>>(x, Wq, bq, Wk, bk, Wv, bv,
                                                       qT, kT, vv);
  attn_kernel<<<nsplit * BATCH * NQB, 256, 0, stream>>>(qT, kT, vv, Opart, Mpart,
                                                        Lpart, tiles_per_split);
  combine_kernel<<<(int)(((size_t)BATCH * CDIM * NDIM) / 256), 256, 0, stream>>>(
      Opart, Mpart, Lpart, out, nsplit);
}
